// ESF_78391743086936
// MI455X (gfx1250) — matrix operand order repaired and checked by reading, not compiled, not run on hardware
//
#include <hip/hip_runtime.h>

typedef __attribute__((ext_vector_type(16))) _Float16 v16h;
typedef __attribute__((ext_vector_type(8)))  _Float16 v8h;
typedef __attribute__((ext_vector_type(8)))  float    v8f;

#define B_TOT 16384
#define NLOC  12
#define FDIM  512
#define HDIM  256
#define MB    16          // batch rows per workgroup
#define NTHREADS 256      // 8 wave32 waves

#define LH_STRIDE 520     // f16 row stride for staged L / G (512 + 8 pad)
#define LF_STRIDE 516     // f32 row stride for lf_mean
#define GQT_STRIDE 20     // f32 stride of transposed gq ([h][m]), 80B -> conflict-free b128
#define CB_STRIDE 528     // f16 row stride for combined (512 + 16 pad)

// f16 element offsets of the swizzled weights in d_ws
#define WQH_OFF 0
#define WKH_OFF (FDIM*HDIM)          // 131072
#define WVH_OFF (2*FDIM*HDIM)        // 262144
#define WFH_OFF (3*FDIM*HDIM)        // 393216, Wf is FDIM x FDIM

struct __align__(16) Smem {
    _Float16 Lh[NLOC * MB * LH_STRIDE];   // 199,680 B  staged local_features (f16, token-major rows)
    _Float16 Gh[MB * LH_STRIDE];          //  16,640 B  staged global_feature (f16)
    float    lf[MB * LF_STRIDE];          //  33,024 B  lf_mean (f32)
    float    gq[HDIM * GQT_STRIDE];       //  20,480 B  transposed gq [h][m] (f32, incl. bq)
    _Float16 comb[MB * CB_STRIDE];        //  16,896 B  [ltg | wg_mean] as f16
    float    s2[NLOC * MB];               //     768 B  logits, token-major [t*16+m]
    float    a2[NLOC * MB];               //     768 B  softmax weights, token-major
};                                        // total 288,256 B < 320 KB

__device__ inline v8f wmma16(v16h a, v16h b, v8f c) {
    return __builtin_amdgcn_wmma_f32_16x16x32_f16(
        false, a, false, b, (short)0, c, false, false);
}

// A fragment from LDS f16 (p -> row start + kstep*32), contiguous K layout:
// lane half 0: K {0..15}; lane half 1: K {16..31}
__device__ inline v16h ldsA(const _Float16* p, int sel) {
    v8h lo = *(const v8h*)(p + sel * 16);
    v8h hi = *(const v8h*)(p + sel * 16 + 8);
    return __builtin_shufflevector(lo, hi, 0,1,2,3,4,5,6,7,8,9,10,11,12,13,14,15);
}

// B fragment from pre-swizzled f16 weights: this lane's 16 contiguous halfs.
__device__ inline v16h ldB(const _Float16* p) {
    v8h lo = *(const v8h*)(p);
    v8h hi = *(const v8h*)(p + 8);
    return __builtin_shufflevector(lo, hi, 0,1,2,3,4,5,6,7,8,9,10,11,12,13,14,15);
}

// --------- pre-pass: f32 weight [K][N] -> f16 fragments (kstep,ntile) of 32 lanes x 16 ---------
__global__ void ESF_swizzle_w(const float* __restrict__ W, _Float16* __restrict__ Wh,
                              int N, int total) {
    int idx = blockIdx.x * blockDim.x + threadIdx.x;
    if (idx >= total) return;
    const int frag  = idx >> 9;         // 512 f16 per fragment
    const int lane  = (idx >> 4) & 31;
    const int i     = idx & 15;
    const int nt    = N >> 4;
    const int kstep = frag / nt;
    const int ntile = frag - kstep * nt;
    const int k = kstep * 32 + (lane >> 4) * 16 + i;   // B layout: half0 K 0..15, half1 K 16..31
    const int n = ntile * 16 + (lane & 15);
    Wh[idx] = (_Float16)W[(size_t)k * N + n];
}

__global__ __launch_bounds__(NTHREADS)
void ESF_fused_attn(const float* __restrict__ G,  const float* __restrict__ L,
                    const _Float16* __restrict__ WqH, const float* __restrict__ bq,
                    const _Float16* __restrict__ WkH,
                    const _Float16* __restrict__ WvH, const float* __restrict__ bv,
                    const float* __restrict__ wdr, const float* __restrict__ bdr,
                    const _Float16* __restrict__ WfH, const float* __restrict__ bf,
                    float* __restrict__ out)
{
    __shared__ Smem sm;
    const int tid    = threadIdx.x;
    const int wave   = tid >> 5;
    const int lane   = tid & 31;
    const int lane16 = lane & 15;
    const int sel    = lane >> 4;
    const int b0     = blockIdx.x * MB;

    float w[NLOC];
#pragma unroll
    for (int t = 0; t < NLOC; t++) w[t] = wdr[t];
    float sw = 0.f;
#pragma unroll
    for (int t = 0; t < NLOC; t++) sw += w[t];
    const float bdr0 = bdr[0];

    v8f vz = {0.f,0.f,0.f,0.f,0.f,0.f,0.f,0.f};

    // ---------- Phase 1: stage L (f16, token-major) + lf_mean (f32) + G (f16) ----------
    for (int u = tid; u < MB * (FDIM / 4); u += NTHREADS) {
        const int bl = u >> 7;          // FDIM/4 == 128
        const int f4 = u & 127;
        const float* src = L + ((size_t)(b0 + bl) * NLOC) * FDIM + f4 * 4;
        float a0 = bdr0, a1 = bdr0, a2v = bdr0, a3 = bdr0;
#pragma unroll
        for (int t = 0; t < NLOC; t++) {
            float4 v = *(const float4*)(src + (size_t)t * FDIM);
            _Float16* dst = &sm.Lh[(t * MB + bl) * LH_STRIDE + f4 * 4];
            dst[0] = (_Float16)v.x; dst[1] = (_Float16)v.y;
            dst[2] = (_Float16)v.z; dst[3] = (_Float16)v.w;
            a0 += w[t] * v.x; a1 += w[t] * v.y; a2v += w[t] * v.z; a3 += w[t] * v.w;
        }
        float* lfp = &sm.lf[bl * LF_STRIDE + f4 * 4];
        lfp[0] = a0; lfp[1] = a1; lfp[2] = a2v; lfp[3] = a3;
        // stage global_feature tile as f16
        float4 g = *(const float4*)(G + (size_t)(b0 + bl) * FDIM + f4 * 4);
        _Float16* gd = &sm.Gh[bl * LH_STRIDE + f4 * 4];
        gd[0] = (_Float16)g.x; gd[1] = (_Float16)g.y;
        gd[2] = (_Float16)g.z; gd[3] = (_Float16)g.w;
    }
    if (tid < NLOC * MB) sm.s2[tid] = 0.f;
    __syncthreads();

    // ---------- Phase 2: gq = G@Wq + bq (transposed store) ; gv -> wg_mean half of combined ----------
    {
        v8f accq[2] = {vz, vz}, accv[2] = {vz, vz};
        for (int ks = 0; ks < 16; ks++) {
            v16h a = ldsA(&sm.Gh[lane16 * LH_STRIDE + ks * 32], sel);
#pragma unroll
            for (int j = 0; j < 2; j++) {
                const int frag = ks * (HDIM/16) + wave * 2 + j;
                accq[j] = wmma16(a, ldB(WqH + ((size_t)frag * 32 + lane) * 16), accq[j]);
                accv[j] = wmma16(a, ldB(WvH + ((size_t)frag * 32 + lane) * 16), accv[j]);
            }
        }
#pragma unroll
        for (int j = 0; j < 2; j++) {
            const int h = (wave * 2 + j) * 16 + lane16;
            const float bqv = bq[h], bvv = bv[h];
            // gq transposed: this lane owns rows m = 8*sel + r (contiguous) of column h
            float4 q0, q1;
            q0.x = accq[j][0] + bqv; q0.y = accq[j][1] + bqv;
            q0.z = accq[j][2] + bqv; q0.w = accq[j][3] + bqv;
            q1.x = accq[j][4] + bqv; q1.y = accq[j][5] + bqv;
            q1.z = accq[j][6] + bqv; q1.w = accq[j][7] + bqv;
            *(float4*)&sm.gq[h * GQT_STRIDE + 8 * sel]     = q0;
            *(float4*)&sm.gq[h * GQT_STRIDE + 8 * sel + 4] = q1;
#pragma unroll
            for (int r = 0; r < 8; r++) {
                const int m = r + 8 * sel;
                sm.comb[m * CB_STRIDE + HDIM + h] =
                    (_Float16)(sw * (accv[j][r] + bvv) + bdr0);
            }
        }
    }
    __syncthreads();

    // ---------- Phase 3: lk = L@Wk, logits s2 = gq . lk  (bk cancels in softmax) ----------
    for (int tc = 0; tc < NLOC; tc += 4) {
        v8f acc[4][2];
#pragma unroll
        for (int tt = 0; tt < 4; tt++) { acc[tt][0] = vz; acc[tt][1] = vz; }
        for (int ks = 0; ks < 16; ks++) {
            v16h a[4];
#pragma unroll
            for (int tt = 0; tt < 4; tt++)
                a[tt] = ldsA(&sm.Lh[((tc + tt) * MB + lane16) * LH_STRIDE + ks * 32], sel);
            if (ks < 15)
                __builtin_prefetch(WkH + ((size_t)((ks+1) * (HDIM/16) + wave*2) * 32 + lane) * 16, 0, 3);
#pragma unroll
            for (int j = 0; j < 2; j++) {
                const int frag = ks * (HDIM/16) + wave * 2 + j;
                v16h bkf = ldB(WkH + ((size_t)frag * 32 + lane) * 16);
#pragma unroll
                for (int tt = 0; tt < 4; tt++)
                    acc[tt][j] = wmma16(a[tt], bkf, acc[tt][j]);
            }
        }
#pragma unroll
        for (int tt = 0; tt < 4; tt++) {
#pragma unroll
            for (int j = 0; j < 2; j++) {
                const int hcol = (wave * 2 + j) * 16 + lane16;
                const float4* gp = (const float4*)&sm.gq[hcol * GQT_STRIDE + 8 * sel];
                float4 g0 = gp[0], g1 = gp[1];
                float p[8];
                p[0] = acc[tt][j][0] * g0.x; p[1] = acc[tt][j][1] * g0.y;
                p[2] = acc[tt][j][2] * g0.z; p[3] = acc[tt][j][3] * g0.w;
                p[4] = acc[tt][j][4] * g1.x; p[5] = acc[tt][j][5] * g1.y;
                p[6] = acc[tt][j][6] * g1.z; p[7] = acc[tt][j][7] * g1.w;
#pragma unroll
                for (int mask = 1; mask < 16; mask <<= 1)
#pragma unroll
                    for (int r = 0; r < 8; r++)
                        p[r] += __shfl_xor(p[r], mask, 32);
                if (lane16 == 0) {
#pragma unroll
                    for (int r = 0; r < 8; r++)
                        atomicAdd(&sm.s2[(tc + tt) * MB + 8 * sel + r], p[r]);
                }
            }
        }
    }
    __syncthreads();

    // ---------- softmax over the 12 tokens (scale = 1/sqrt(256)) ----------
    if (tid < MB) {
        float sv[NLOC], mx = -1e30f;
#pragma unroll
        for (int t = 0; t < NLOC; t++) {
            sv[t] = sm.s2[t * MB + tid] * 0.0625f;
            mx = fmaxf(mx, sv[t]);
        }
        float z = 0.f;
#pragma unroll
        for (int t = 0; t < NLOC; t++) { sv[t] = __expf(sv[t] - mx); z += sv[t]; }
        const float inv = 1.f / z;
#pragma unroll
        for (int t = 0; t < NLOC; t++) sm.a2[t * MB + tid] = sv[t] * inv;
    }
    __syncthreads();

    // ---------- Phase 4: lv = L@Wv, ltg = sum_t a2*lv (+bv) -> low half of combined ----------
    {
        v8f ltg[2] = {vz, vz};
        for (int tc = 0; tc < NLOC; tc += 4) {
            v8f acc[4][2];
#pragma unroll
            for (int tt = 0; tt < 4; tt++) { acc[tt][0] = vz; acc[tt][1] = vz; }
            for (int ks = 0; ks < 16; ks++) {
                v16h a[4];
#pragma unroll
                for (int tt = 0; tt < 4; tt++)
                    a[tt] = ldsA(&sm.Lh[((tc + tt) * MB + lane16) * LH_STRIDE + ks * 32], sel);
                if (ks < 15)
                    __builtin_prefetch(WvH + ((size_t)((ks+1) * (HDIM/16) + wave*2) * 32 + lane) * 16, 0, 3);
#pragma unroll
                for (int j = 0; j < 2; j++) {
                    const int frag = ks * (HDIM/16) + wave * 2 + j;
                    v16h bvf = ldB(WvH + ((size_t)frag * 32 + lane) * 16);
#pragma unroll
                    for (int tt = 0; tt < 4; tt++)
                        acc[tt][j] = wmma16(a[tt], bvf, acc[tt][j]);
                }
            }
#pragma unroll
            for (int tt = 0; tt < 4; tt++) {
                const float4* ap = (const float4*)&sm.a2[(tc + tt) * MB + 8 * sel];
                float4 a0 = ap[0], a1 = ap[1];
#pragma unroll
                for (int j = 0; j < 2; j++) {
                    ltg[j][0] += a0.x * acc[tt][j][0]; ltg[j][1] += a0.y * acc[tt][j][1];
                    ltg[j][2] += a0.z * acc[tt][j][2]; ltg[j][3] += a0.w * acc[tt][j][3];
                    ltg[j][4] += a1.x * acc[tt][j][4]; ltg[j][5] += a1.y * acc[tt][j][5];
                    ltg[j][6] += a1.z * acc[tt][j][6]; ltg[j][7] += a1.w * acc[tt][j][7];
                }
            }
        }
#pragma unroll
        for (int j = 0; j < 2; j++) {
            const int h = (wave * 2 + j) * 16 + lane16;
            const float bvv = bv[h];
#pragma unroll
            for (int r = 0; r < 8; r++)
                sm.comb[(r + 8 * sel) * CB_STRIDE + h] = (_Float16)(ltg[j][r] + bvv);
        }
    }
    __syncthreads();

    // ---------- Phase 5: out = relu(comb@Wf + bf) + lf_mean + G ----------
    {
        v8f acco[4] = {vz, vz, vz, vz};
        const _Float16* crow = &sm.comb[lane16 * CB_STRIDE];
        for (int ks = 0; ks < 16; ks++) {
            v16h a = ldsA(crow + ks * 32, sel);
            if (ks < 15)
                __builtin_prefetch(WfH + ((size_t)((ks+1) * (FDIM/16) + wave*4) * 32 + lane) * 16, 0, 3);
#pragma unroll
            for (int jj = 0; jj < 4; jj++) {
                const int frag = ks * (FDIM/16) + wave * 4 + jj;
                acco[jj] = wmma16(a, ldB(WfH + ((size_t)frag * 32 + lane) * 16), acco[jj]);
            }
        }
#pragma unroll
        for (int jj = 0; jj < 4; jj++) {
            const int ng = (wave * 4 + jj) * 16 + lane16;
            const float bfv = bf[ng];
#pragma unroll
            for (int r = 0; r < 8; r++) {
                const int m = r + 8 * sel;
                float v = fmaxf(acco[jj][r] + bfv, 0.f);
                v += sm.lf[m * LF_STRIDE + ng] + G[(size_t)(b0 + m) * FDIM + ng];
                out[(size_t)(b0 + m) * FDIM + ng] = v;
            }
        }
    }
}

extern "C" void kernel_launch(void* const* d_in, const int* in_sizes, int n_in,
                              void* d_out, int out_size, void* d_ws, size_t ws_size,
                              hipStream_t stream) {
    (void)in_sizes; (void)n_in; (void)out_size; (void)ws_size;
    const float* G   = (const float*)d_in[0];
    const float* L   = (const float*)d_in[1];
    const float* Wq  = (const float*)d_in[2];
    const float* bq  = (const float*)d_in[3];
    const float* Wk  = (const float*)d_in[4];
    const float* Wv  = (const float*)d_in[6];
    const float* bv  = (const float*)d_in[7];
    const float* wdr = (const float*)d_in[8];
    const float* bdr = (const float*)d_in[9];
    const float* Wf  = (const float*)d_in[10];
    const float* bf  = (const float*)d_in[11];
    float* out = (float*)d_out;

    _Float16* wsh = (_Float16*)d_ws;     // needs 1,310,720 B of workspace
    _Float16* WqH = wsh + WQH_OFF;
    _Float16* WkH = wsh + WKH_OFF;
    _Float16* WvH = wsh + WVH_OFF;
    _Float16* WfH = wsh + WFH_OFF;

    const int nQKV = FDIM * HDIM;        // 131072
    const int nF   = FDIM * FDIM;        // 262144
    ESF_swizzle_w<<<dim3(nQKV / 256), dim3(256), 0, stream>>>(Wq, WqH, HDIM, nQKV);
    ESF_swizzle_w<<<dim3(nQKV / 256), dim3(256), 0, stream>>>(Wk, WkH, HDIM, nQKV);
    ESF_swizzle_w<<<dim3(nQKV / 256), dim3(256), 0, stream>>>(Wv, WvH, HDIM, nQKV);
    ESF_swizzle_w<<<dim3(nF   / 256), dim3(256), 0, stream>>>(Wf, WfH, FDIM, nF);

    ESF_fused_attn<<<dim3(B_TOT / MB), dim3(NTHREADS), 0, stream>>>(
        G, L, WqH, bq, WkH, WvH, bv, wdr, bdr, WfH, bf, out);
}
